// Encoding_25735444037928
// MI455X (gfx1250) — matrix operand order repaired and checked by reading, not compiled, not run on hardware
//
#include <hip/hip_runtime.h>
#include <hip/hip_bf16.h>
#include <math.h>

// Problem constants (from reference): B=16, P_LEN=1024, D=512
#define BB  16
#define LL  1024
#define DD  512
#define DC  1024   // concat dim

typedef __attribute__((ext_vector_type(16))) __bf16 v16bf;
typedef __attribute__((ext_vector_type(8)))  __bf16 v8bf;
typedef __attribute__((ext_vector_type(8)))  float  v8f;

// ---------------------------------------------------------------------------
// CDNA5 async copy: global -> LDS, tracked by ASYNCcnt (ISA ch.10 / §15.18,
// opcode 98 GLOBAL_LOAD_ASYNC_TO_LDS_B128). vdst = VGPR with LDS byte addr,
// vaddr = 64-bit global address. Low 32 bits of a generic LDS pointer are the
// LDS offset (aperture check uses only addr[63:32]).
// ---------------------------------------------------------------------------
__device__ __forceinline__ void async_load_b128(const void* smem_ptr,
                                                const void* gaddr) {
  unsigned lds = (unsigned)(unsigned long long)smem_ptr;
  asm volatile("global_load_async_to_lds_b128 %0, %1, off"
               :: "v"(lds), "v"(gaddr)
               : "memory");
}
#define S_WAIT_ASYNC(n) asm volatile("s_wait_asynccnt " #n ::: "memory")

// ---------------------------------------------------------------------------
// WMMA helper (v_wmma_f32_16x16x32_bf16, D = A(16x32) * B(32x16) + C)
// ---------------------------------------------------------------------------
__device__ __forceinline__ v8f wmma_bf16(v16bf a, v16bf b, v8f c) {
  return __builtin_amdgcn_wmma_f32_16x16x32_bf16(
      /*neg_a=*/false, a, /*neg_b=*/false, b,
      /*c_mod=*/(short)0, c, /*reuse_a=*/false, /*reuse_b=*/false);
}

// A-matrix 16x32 bf16 fragment. Per-lane element e holds
//   K = (lane>>4)*16 + e,  M = lane&15.
// Source is row-major [row][k] with given stride; two contiguous 8-elem runs.
__device__ __forceinline__ v16bf load_a_frag(const __bf16* base, int stride,
                                             int row0, int k0, int lane) {
  const int m = lane & 15, g = lane >> 4;
  const __bf16* p = base + (size_t)(row0 + m) * stride + k0 + g * 16;
  v8bf lo = *(const v8bf*)p;          // e0..7  -> K = k0 + g*16 + idx
  v8bf hi = *(const v8bf*)(p + 8);    // e8..15 -> K = k0 + g*16 + 8 + idx
  v16bf a;
#pragma unroll
  for (int i = 0; i < 8; ++i) { a[i] = lo[i]; a[8 + i] = hi[i]; }
  return a;
}

// B fragment from an LDS-staged [n][32] bf16 tile (row = N, 32-elem K run).
// Per-lane element e holds K = (lane>>4)*16 + e, N = n0 + (lane&15).
__device__ __forceinline__ v16bf load_b_frag_lds(const __bf16* tile, int n0,
                                                 int lane) {
  const int n = lane & 15, g = lane >> 4;
  return *(const v16bf*)(tile + (n0 + n) * 32 + g * 16);
}

// ---------------------------------------------------------------------------
// Prep kernels: bf16 conversions, transposes, sj = P @ wb
// ---------------------------------------------------------------------------
__global__ void prep_convert(const float* __restrict__ P,
                             const float* __restrict__ watt,
                             __bf16* __restrict__ Pb,   // [B*L, D]
                             __bf16* __restrict__ Qb,   // [B*L, D]  P*wc
                             __bf16* __restrict__ PT) { // [B, D, L] transpose
  size_t idx = (size_t)blockIdx.x * blockDim.x + threadIdx.x;
  if (idx >= (size_t)BB * LL * DD) return;
  int d = (int)(idx % DD);
  size_t bi = idx / DD;                    // b*L + i
  int b = (int)(bi / LL), i = (int)(bi % LL);
  float p = P[idx];
  __bf16 pb = (__bf16)p;
  Pb[idx] = pb;
  Qb[idx] = (__bf16)(p * watt[2 * DD + d]);
  PT[((size_t)b * DD + d) * LL + i] = pb;
}

__global__ void prep_w(const float* __restrict__ w1,
                       const float* __restrict__ w2,
                       const float* __restrict__ w3,
                       __bf16* __restrict__ WT) {  // 3 x [D, DC] (n-major)
  int idx = blockIdx.x * blockDim.x + threadIdx.x;  // over DC*D
  if (idx >= DC * DD) return;
  int k = idx / DD, n = idx % DD;
  size_t o = (size_t)n * DC + k;
  WT[o]                        = (__bf16)w1[idx];
  WT[(size_t)DD * DC + o]      = (__bf16)w2[idx];
  WT[2 * (size_t)DD * DC + o]  = (__bf16)w3[idx];
}

__global__ void prep_sj(const float* __restrict__ P,
                        const float* __restrict__ watt,
                        float* __restrict__ sj) {   // [B*L]
  int row = blockIdx.x * blockDim.x + threadIdx.x;
  if (row >= BB * LL) return;
  const float* p = P + (size_t)row * DD;
  float s = 0.f;
  for (int d = 0; d < DD; ++d) s += p[d] * watt[DD + d];
  sj[row] = s;
}

// ---------------------------------------------------------------------------
// Logits: A[b][i][j] = sum_d Q[b,i,d]*P[b,j,d] + sj[b,j]   (si is constant
// along the softmax axis so it cancels and is dropped).
// 8 waves/WG; wave = 16 i-rows x 128 j-cols. The 128x32 B tile is shared by
// all 8 waves: staged once per WG into LDS with async loads (double-buffered,
// ASYNCcnt-tracked). B-frag ds reads are software-pipelined one WMMA ahead.
// ---------------------------------------------------------------------------
__global__ void logits_kernel(const __bf16* __restrict__ Q,
                              const __bf16* __restrict__ Pb,
                              const float* __restrict__ sj,
                              float* __restrict__ logits) {
  __shared__ __align__(128) __bf16 smem[2][128 * 32];  // 2 x 8 KiB
  const int tid = threadIdx.x;
  const int lane = tid & 31;
  const int wave = tid >> 5;
  const int b = blockIdx.z;
  const int iBase = blockIdx.x * 128 + wave * 16;
  const int jBase = blockIdx.y * 128;
  const __bf16* Qb  = Q  + (size_t)b * LL * DD;
  const __bf16* Pbb = Pb + (size_t)b * LL * DD;

  // Stage 128 rows (j) x 32 cols (k) of P into smem[buf]: 512 16-B chunks.
  auto issue = [&](int buf, int k0) {
#pragma unroll
    for (int r = 0; r < 2; ++r) {
      int c = tid + 256 * r;        // chunk id 0..511
      int row = c >> 2;             // 4 chunks per 64-B row
      int col = (c & 3) * 8;        // element offset (8 bf16 = 16 B)
      async_load_b128(&smem[buf][row * 32 + col],
                      Pbb + (size_t)(jBase + row) * DD + k0 + col);
    }
  };

  issue(0, 0);
  v8f acc[8] = {};
  for (int kt = 0; kt < DD / 32; ++kt) {
    const int cur = kt & 1;
    if (kt + 1 < DD / 32) { issue(cur ^ 1, (kt + 1) * 32); S_WAIT_ASYNC(2); }
    else                  { S_WAIT_ASYNC(0); }
    __syncthreads();                          // cur tile visible to all waves
    v16bf a = load_a_frag(Qb, DD, iBase, kt * 32, lane);
    v16bf bcur = load_b_frag_lds(smem[cur], 0, lane);
#pragma unroll
    for (int jt = 0; jt < 8; ++jt) {
      v16bf bnext = (jt < 7) ? load_b_frag_lds(smem[cur], (jt + 1) * 16, lane)
                             : bcur;
      acc[jt] = wmma_bf16(a, bcur, acc[jt]);
      bcur = bnext;
    }
    __syncthreads();                          // WAR: done reading before refill
  }

  const int m0 = (lane >> 4) * 8, n = lane & 15;
#pragma unroll
  for (int jt = 0; jt < 8; ++jt) {
    int j = jBase + jt * 16 + n;
    float sjv = sj[b * LL + j];
#pragma unroll
    for (int rr = 0; rr < 8; ++rr) {
      int i = iBase + m0 + rr;
      logits[((size_t)b * LL + i) * LL + j] = acc[jt][rr] + sjv;
    }
  }
}

// ---------------------------------------------------------------------------
// Softmax over axis j (1024 wide): one wave32 per row, shuffle reductions.
// ---------------------------------------------------------------------------
__global__ void softmax_kernel(const float* __restrict__ logits,
                               __bf16* __restrict__ SA) {
  const int lane = threadIdx.x & 31;
  const int wave = threadIdx.x >> 5;
  const size_t row = (size_t)blockIdx.x * 8 + wave;  // over B*L rows
  const float* src = logits + row * LL;
  float x[32];
  float m = -1e30f;
#pragma unroll
  for (int t = 0; t < 32; ++t) { x[t] = src[t * 32 + lane]; m = fmaxf(m, x[t]); }
#pragma unroll
  for (int off = 16; off >= 1; off >>= 1) m = fmaxf(m, __shfl_xor(m, off, 32));
  float s = 0.f;
#pragma unroll
  for (int t = 0; t < 32; ++t) { x[t] = __expf(x[t] - m); s += x[t]; }
#pragma unroll
  for (int off = 16; off >= 1; off >>= 1) s += __shfl_xor(s, off, 32);
  const float inv = 1.f / s;
  __bf16* dst = SA + row * LL;
#pragma unroll
  for (int t = 0; t < 32; ++t) dst[t * 32 + lane] = (__bf16)(x[t] * inv);
}

// ---------------------------------------------------------------------------
// attn[b,i,d] = sum_j SA[b,i,j] * P[b,j,d]; B-operand from PT[b][d][j] so the
// K run is contiguous. Same LDS async double-buffer staging as logits.
// Wave = 16 i-rows x 128 d-cols.
// ---------------------------------------------------------------------------
__global__ void attn_kernel(const __bf16* __restrict__ SA,
                            const __bf16* __restrict__ PT,
                            __bf16* __restrict__ Attn) {
  __shared__ __align__(128) __bf16 smem[2][128 * 32];
  const int tid = threadIdx.x;
  const int lane = tid & 31;
  const int wave = tid >> 5;
  const int b = blockIdx.z;
  const int iBase = blockIdx.x * 128 + wave * 16;
  const int dBase = blockIdx.y * 128;
  const __bf16* SAb = SA + (size_t)b * LL * LL;
  const __bf16* PTb = PT + (size_t)b * DD * LL;

  auto issue = [&](int buf, int j0) {
#pragma unroll
    for (int r = 0; r < 2; ++r) {
      int c = tid + 256 * r;
      int row = c >> 2;
      int col = (c & 3) * 8;
      async_load_b128(&smem[buf][row * 32 + col],
                      PTb + (size_t)(dBase + row) * LL + j0 + col);
    }
  };

  issue(0, 0);
  v8f acc[8] = {};
  for (int jt = 0; jt < LL / 32; ++jt) {
    const int cur = jt & 1;
    if (jt + 1 < LL / 32) { issue(cur ^ 1, (jt + 1) * 32); S_WAIT_ASYNC(2); }
    else                  { S_WAIT_ASYNC(0); }
    __syncthreads();
    v16bf a = load_a_frag(SAb, LL, iBase, jt * 32, lane);
    v16bf bcur = load_b_frag_lds(smem[cur], 0, lane);
#pragma unroll
    for (int nt = 0; nt < 8; ++nt) {
      v16bf bnext = (nt < 7) ? load_b_frag_lds(smem[cur], (nt + 1) * 16, lane)
                             : bcur;
      acc[nt] = wmma_bf16(a, bcur, acc[nt]);
      bcur = bnext;
    }
    __syncthreads();
  }

  const int m0 = (lane >> 4) * 8, n = lane & 15;
#pragma unroll
  for (int nt = 0; nt < 8; ++nt)
#pragma unroll
    for (int rr = 0; rr < 8; ++rr) {
      int i = iBase + m0 + rr, d = dBase + nt * 16 + n;
      Attn[((size_t)b * LL + i) * DD + d] = (__bf16)acc[nt][rr];
    }
}

// ---------------------------------------------------------------------------
// FFN: Pc = [P, attn] (virtual concat; k-chunks of 32 never straddle 512).
// WG = 128 rows (8 waves x 16) sharing one 64-col range; the 3x64x32 weight
// tile (12 KiB) is async-staged into LDS (double-buffered) so weight traffic
// is paid once per WG instead of once per wave (8x L2-read reduction).
// out = sigmoid(Pc@w2+b2)*P + sigmoid(Pc@w3+b3)*tanh(Pc@w1+b1)
// ---------------------------------------------------------------------------
__global__ void ffn_kernel(const __bf16* __restrict__ Pb,
                           const __bf16* __restrict__ Attn,
                           const __bf16* __restrict__ WT,
                           const float* __restrict__ b1,
                           const float* __restrict__ b2,
                           const float* __restrict__ b3,
                           const float* __restrict__ Pf,
                           float* __restrict__ out) {
  __shared__ __align__(128) __bf16 smem[2][3 * 64 * 32];  // 2 x 12 KiB
  const int tid = threadIdx.x;
  const int lane = tid & 31;
  const int wave = tid >> 5;
  const int rowBase = blockIdx.x * 128 + wave * 16;   // over B*L rows
  const int colBase = blockIdx.y * 64;                // over D cols (WG-wide)
  const __bf16* WTs[3] = {WT, WT + (size_t)DD * DC, WT + 2 * (size_t)DD * DC};

  // Stage [3 gates][64 n-rows][32 k] of WT: 768 16-B chunks, 3 per thread.
  auto issue = [&](int buf, int k0) {
#pragma unroll
    for (int r = 0; r < 3; ++r) {
      int c = tid + 256 * r;        // chunk id 0..767
      int nf = c >> 2;              // flat (g*64+n) row, 4 chunks per 64-B row
      int g = nf >> 6;
      int n = nf & 63;
      int col = (c & 3) * 8;
      async_load_b128(&smem[buf][nf * 32 + col],
                      WTs[g] + (size_t)(colBase + n) * DC + k0 + col);
    }
  };

  issue(0, 0);
  v8f acc[3][4] = {};
  for (int kt = 0; kt < DC / 32; ++kt) {
    const int cur = kt & 1;
    const int k0 = kt * 32;
    if (kt + 1 < DC / 32) { issue(cur ^ 1, (kt + 1) * 32); S_WAIT_ASYNC(3); }
    else                  { S_WAIT_ASYNC(0); }
    __syncthreads();
    const __bf16* src = (k0 < DD) ? Pb : Attn;
    const int kk = (k0 < DD) ? k0 : (k0 - DD);
    v16bf a = load_a_frag(src, DD, rowBase, kk, lane);
    // prefetch next k-step's A-row bytes (global_prefetch_b8)
    if (k0 + 32 < DC) {
      const __bf16* nsrc = (k0 + 32 < DD) ? Pb : Attn;
      const int nkk = (k0 + 32 < DD) ? (k0 + 32) : (k0 + 32 - DD);
      __builtin_prefetch(nsrc + (size_t)(rowBase + (lane & 15)) * DD + nkk, 0, 1);
    }
    v16bf bcur = load_b_frag_lds(smem[cur], 0, lane);  // g=0, nt=0
#pragma unroll
    for (int nt = 0; nt < 4; ++nt)
#pragma unroll
      for (int g = 0; g < 3; ++g) {
        const int q = nt * 3 + g;
        v16bf bnext;
        if (q < 11) {
          const int qn = q + 1, ntn = qn / 3, gn = qn % 3;
          bnext = load_b_frag_lds(smem[cur] + gn * 64 * 32, ntn * 16, lane);
        } else {
          bnext = bcur;
        }
        acc[g][nt] = wmma_bf16(a, bcur, acc[g][nt]);
        bcur = bnext;
      }
    __syncthreads();
  }

  const int m0 = (lane >> 4) * 8, n = lane & 15;
#pragma unroll
  for (int nt = 0; nt < 4; ++nt) {
    int col = colBase + nt * 16 + n;
    float bb1 = b1[col], bb2 = b2[col], bb3 = b3[col];
#pragma unroll
    for (int rr = 0; rr < 8; ++rr) {
      int row = rowBase + m0 + rr;
      float z = tanhf(acc[0][nt][rr] + bb1);
      float r = 1.f / (1.f + __expf(-(acc[1][nt][rr] + bb2)));
      float f = 1.f / (1.f + __expf(-(acc[2][nt][rr] + bb3)));
      float pv = Pf[(size_t)row * DD + col];
      out[(size_t)row * DD + col] = r * pv + f * z;
    }
  }
}

// ---------------------------------------------------------------------------
// Launch: workspace layout (~163 MiB; logits + probs stay L2-resident)
// ---------------------------------------------------------------------------
extern "C" void kernel_launch(void* const* d_in, const int* in_sizes, int n_in,
                              void* d_out, int out_size, void* d_ws, size_t ws_size,
                              hipStream_t stream) {
  const float* P     = (const float*)d_in[0];
  const float* watt  = (const float*)d_in[1];
  const float* w1    = (const float*)d_in[2];
  const float* w2    = (const float*)d_in[3];
  const float* w3    = (const float*)d_in[4];
  const float* b1    = (const float*)d_in[5];
  const float* b2    = (const float*)d_in[6];
  const float* b3    = (const float*)d_in[7];
  float* out = (float*)d_out;

  char* w = (char*)d_ws;
  const size_t SZ_P16 = (size_t)BB * LL * DD * 2;   // 16 MiB
  __bf16* Pb   = (__bf16*)(w);                      size_t off = SZ_P16;
  __bf16* Qb   = (__bf16*)(w + off);                off += SZ_P16;
  __bf16* PT   = (__bf16*)(w + off);                off += SZ_P16;
  __bf16* Attn = (__bf16*)(w + off);                off += SZ_P16;
  __bf16* WT   = (__bf16*)(w + off);                off += 3 * (size_t)DD * DC * 2;
  float*  sj   = (float*)(w + off);                 off += (size_t)BB * LL * 4;
  off = (off + 255) & ~(size_t)255;
  float*  logits = (float*)(w + off);               off += (size_t)BB * LL * LL * 4;
  __bf16* SA   = (__bf16*)(w + off);                off += (size_t)BB * LL * LL * 2;
  (void)ws_size; (void)in_sizes; (void)n_in; (void)out_size;

  // 1) conversions / transposes
  {
    size_t n = (size_t)BB * LL * DD;
    prep_convert<<<dim3((unsigned)((n + 255) / 256)), dim3(256), 0, stream>>>(
        P, watt, Pb, Qb, PT);
  }
  prep_w<<<dim3((DC * DD + 255) / 256), dim3(256), 0, stream>>>(w1, w2, w3, WT);
  prep_sj<<<dim3((BB * LL + 255) / 256), dim3(256), 0, stream>>>(P, watt, sj);

  // 2) logits = Q @ P^T + sj
  logits_kernel<<<dim3(LL / 128, LL / 128, BB), dim3(256), 0, stream>>>(
      Qb, Pb, sj, logits);

  // 3) row softmax -> bf16 probs
  softmax_kernel<<<dim3(BB * LL / 8), dim3(256), 0, stream>>>(logits, SA);

  // 4) attn = SA @ P
  attn_kernel<<<dim3(LL / 128, DD / 128, BB), dim3(256), 0, stream>>>(SA, PT, Attn);

  // 5) FFN + gating (weights LDS-staged, 8x L2 traffic reduction)
  ffn_kernel<<<dim3(BB * LL / 128, DD / 64), dim3(256), 0, stream>>>(
      Pb, Attn, WT, b1, b2, b3, P, out);
}
